// MultiHeadAttention_72395968741781
// MI455X (gfx1250) — matrix operand order repaired and checked by reading, not compiled, not run on hardware
//
#include <hip/hip_runtime.h>

typedef __bf16 bf16_t;
typedef __attribute__((ext_vector_type(16))) __bf16 v16bf;
typedef __attribute__((ext_vector_type(8)))  __bf16 v8bf;
typedef __attribute__((ext_vector_type(8)))  float  v8f;

#if __has_builtin(__builtin_amdgcn_global_load_async_to_lds_b128)
#define HAVE_ASYNC_LDS 1
#else
#define HAVE_ASYNC_LDS 0
#endif

__device__ inline v16bf join16(v8bf lo, v8bf hi) {
  return __builtin_shufflevector(lo, hi, 0,1,2,3,4,5,6,7,8,9,10,11,12,13,14,15);
}

__device__ inline v8f wmma_bf16(v16bf a, v16bf b, v8f c) {
  // (neg_a, A, neg_b, B, c_mod, C, reuse_a, reuse_b)
  return __builtin_amdgcn_wmma_f32_16x16x32_bf16(false, a, false, b, (short)0, c, false, false);
}

__device__ inline bf16_t f2bf(float f) { return static_cast<bf16_t>(f); }

#if HAVE_ASYNC_LDS
typedef int v4i_vs __attribute__((vector_size(16)));
__device__ inline void async_ld_b128(const bf16_t* g, bf16_t* l) {
  __builtin_amdgcn_global_load_async_to_lds_b128(
      (__attribute__((address_space(1))) v4i_vs*)(v4i_vs*)(void*)const_cast<bf16_t*>(g),
      (__attribute__((address_space(3))) v4i_vs*)(v4i_vs*)(void*)l,
      0, 0);
}
#endif

__device__ inline void wait_async0() {
#if __has_builtin(__builtin_amdgcn_s_wait_asynccnt)
  __builtin_amdgcn_s_wait_asynccnt(0);
#else
  asm volatile("s_wait_asynccnt 0x0" ::: "memory");
#endif
}

// ---- A-fragment (16x32 bf16, MxK) from global row-major, ld elements ----
__device__ inline v16bf load_afrag_g(const bf16_t* __restrict__ base, int ld,
                                     int row0, int k0, int lane) {
  int m = lane & 15, half = lane >> 4;
  const bf16_t* p = base + (size_t)(row0 + m) * ld + k0 + half * 8;
  v8bf lo = *(const v8bf*)p;
  v8bf hi = *(const v8bf*)(p + 16);
  return join16(lo, hi);
}
// ---- A-fragment from LDS tile (row stride 40 bf16, K=32 tile) ----
__device__ inline v16bf lds_afrag(const bf16_t* buf, int row0, int lane) {
  int m = lane & 15, half = lane >> 4;
  const bf16_t* p = buf + (row0 + m) * 40 + half * 8;
  v8bf lo = *(const v8bf*)p;
  v8bf hi = *(const v8bf*)(p + 16);
  return join16(lo, hi);
}
// ---- B-fragment (32x16, KxN) from LDS tile stored [N][K] with row stride ----
__device__ inline v16bf lds_bfrag(const bf16_t* buf, int stride, int row0, int k0, int lane) {
  int n = lane & 15, khalf = lane >> 4;
  const bf16_t* p = buf + (row0 + n) * stride + k0 + khalf * 8;
  v8bf lo = *(const v8bf*)p;
  v8bf hi = *(const v8bf*)(p + 16);
  return join16(lo, hi);
}

// ================= fp32 -> bf16 conversion =================
__global__ void __launch_bounds__(256)
cvt_f32_to_bf16(const float* __restrict__ src, bf16_t* __restrict__ dst, int n) {
  int i = (blockIdx.x * blockDim.x + threadIdx.x) * 4;
  if (i + 3 < n) {
    float4 f = *(const float4*)(src + i);
    dst[i + 0] = f2bf(f.x); dst[i + 1] = f2bf(f.y);
    dst[i + 2] = f2bf(f.z); dst[i + 3] = f2bf(f.w);
  } else {
    for (; i < n; ++i) dst[i] = f2bf(src[i]);
  }
}

// ================= tiled GEMM core =================
// Block: 256 threads / 8 waves as 2(M) x 4(N); block tile 128x256, wave 64x64.
// K-step 32, double-buffered LDS staging (async if available).
struct StageG { v8bf ra, rb0, rb1; };

__device__ inline void stageg_issue(const bf16_t* __restrict__ Ag,
                                    const bf16_t* __restrict__ Bg,
                                    int tmb, int tnb, int k0,
                                    bf16_t* An, bf16_t* Bn, int t, StageG& r) {
  const int arow = t >> 1, acol = (t & 1) * 16;           // A: 128 rows x 32
  const bf16_t* ga  = Ag + (size_t)(tmb + arow) * 1024 + k0 + acol;
  const bf16_t* gb0 = Bg + (size_t)(tnb + t) * 1024 + k0; // B: 256 rows x 32
#if HAVE_ASYNC_LDS
  async_ld_b128(ga,  An + arow * 40 + acol);
  async_ld_b128(gb0,      Bn + t * 40);
  async_ld_b128(gb0 + 16, Bn + t * 40 + 16);
  (void)r;
#else
  r.ra  = *(const v8bf*)ga;
  r.rb0 = *(const v8bf*)gb0;
  r.rb1 = *(const v8bf*)(gb0 + 16);
#endif
}
__device__ inline void stageg_commit(bf16_t* An, bf16_t* Bn, int t, const StageG& r) {
#if HAVE_ASYNC_LDS
  (void)An; (void)Bn; (void)t; (void)r;
  wait_async0();
#else
  const int arow = t >> 1, acol = (t & 1) * 16;
  *(v8bf*)(An + arow * 40 + acol) = r.ra;
  *(v8bf*)(Bn + t * 40)           = r.rb0;
  *(v8bf*)(Bn + t * 40 + 16)      = r.rb1;
#endif
}

__device__ inline void gemm_core_tiled(const bf16_t* __restrict__ Ag,
                                       const bf16_t* __restrict__ Bg,
                                       bf16_t* At, bf16_t* Bt,   // [2][128][40], [2][256][40]
                                       int tmb, int tnb, int wm, int wn,
                                       int lane, int t, v8f acc[4][4]) {
  StageG sr;
  bf16_t* Ac = At;              bf16_t* Bc = Bt;
  bf16_t* An = At + 128 * 40;   bf16_t* Bn = Bt + 256 * 40;
  stageg_issue(Ag, Bg, tmb, tnb, 0, Ac, Bc, t, sr);
  stageg_commit(Ac, Bc, t, sr);
  __syncthreads();
#pragma unroll 1
  for (int kk = 0; kk < 32; ++kk) {
    if (kk < 31) stageg_issue(Ag, Bg, tmb, tnb, (kk + 1) * 32, An, Bn, t, sr);
    v16bf a[4];
#pragma unroll
    for (int i = 0; i < 4; ++i) a[i] = lds_afrag(Ac, wm + 16 * i, lane);
#pragma unroll
    for (int j = 0; j < 4; ++j) {
      v16bf b = lds_bfrag(Bc, 40, wn + 16 * j, 0, lane);
#pragma unroll
      for (int i = 0; i < 4; ++i) acc[i][j] = wmma_bf16(a[i], b, acc[i][j]);
    }
    if (kk < 31) stageg_commit(An, Bn, t, sr);
    __syncthreads();
    bf16_t* tp;
    tp = Ac; Ac = An; An = tp;
    tp = Bc; Bc = Bn; Bn = tp;
  }
}

// ================= fused Q/K/V projection =================
__global__ void __launch_bounds__(256)
qkv_proj(const bf16_t* __restrict__ xb,
         const bf16_t* __restrict__ wq, const bf16_t* __restrict__ wk,
         const bf16_t* __restrict__ wv,
         const float* __restrict__ bq, const float* __restrict__ bk,
         const float* __restrict__ bv,
         bf16_t* __restrict__ Qb, bf16_t* __restrict__ Kb,
         bf16_t* __restrict__ Vt) {
  __shared__ bf16_t Atile[2][128][40];
  __shared__ bf16_t Btile[2][256][40];

  const int t = threadIdx.x;
  const int lane = t & 31, wave = t >> 5;
  const int which = blockIdx.z;
  const bf16_t* W   = (which == 0) ? wq : (which == 1) ? wk : wv;
  const float* bias = (which == 0) ? bq : (which == 1) ? bk : bv;

  const int tmb = (blockIdx.x >> 2) * 128;        // 32 M-blocks
  const int tnb = (blockIdx.x & 3) * 256;         // 4 N-blocks
  const int wm = (wave >> 2) * 64;                // wave tile inside block
  const int wn = (wave & 3) * 64;

  v8f acc[4][4] = {};
  gemm_core_tiled(xb, W, &Atile[0][0][0], &Btile[0][0][0], tmb, tnb, wm, wn, lane, t, acc);

  const int tm = tmb + wm, tn = tnb + wn;
  const int half = lane >> 4, c = lane & 15;
#pragma unroll
  for (int i = 0; i < 4; ++i) {
#pragma unroll
    for (int j = 0; j < 4; ++j) {
      const int ncol = tn + 16 * j + c;
      const float bval = bias[ncol];
      const int h = ncol >> 6, d = ncol & 63;
      if (which == 2) {
        // V transposed [b,h,dh,s]: 8 consecutive s per lane -> one b128 store
        const int mbase = tm + 16 * i + 8 * half;
        const int bidx = mbase >> 10, s0 = mbase & 1023;
        v8bf pk;
#pragma unroll
        for (int v = 0; v < 8; ++v) pk[v] = f2bf(acc[i][j][v] + bval);
        *(v8bf*)(Vt + (((size_t)(bidx * 16 + h)) * 64 + d) * 1024 + s0) = pk;
      } else {
#pragma unroll
        for (int v = 0; v < 8; ++v) {
          const int mrow = tm + 16 * i + v + 8 * half;
          const int bidx = mrow >> 10, s = mrow & 1023;
          const float val = acc[i][j][v] + bval;
          if (which == 0)
            Qb[(((size_t)(bidx * 16 + h)) * 1024 + s) * 64 + d] = f2bf(val * 0.125f);
          else
            Kb[(((size_t)(bidx * 16 + h)) * 1024 + s) * 64 + d] = f2bf(val);
        }
      }
    }
  }
}

// ================= flash attention =================
// Block: 8 waves = 128 q rows of one (b,h); K/V kv-tiles (32 wide) staged in
// LDS, shared by all waves; double-buffered; online softmax per wave.
struct StageA { v8bf r; };

__device__ inline void stagea_issue(const bf16_t* __restrict__ Kp,
                                    const bf16_t* __restrict__ Vp, int kvn,
                                    bf16_t* Kn, bf16_t* Vn, int t, StageA& s) {
  if (t < 128) {                       // K tile: 32 rows x 64 (stride 72)
    const int row = t >> 2, c = (t & 3) * 16;
    const bf16_t* g = Kp + (size_t)(kvn + row) * 64 + c;
#if HAVE_ASYNC_LDS
    async_ld_b128(g, Kn + row * 72 + c);
#else
    s.r = *(const v8bf*)g;
#endif
  } else {                             // V tile: 64 rows x 32 (stride 40)
    const int tt = t - 128;
    const int row = tt >> 1, c = (tt & 1) * 16;
    const bf16_t* g = Vp + (size_t)row * 1024 + kvn + c;
#if HAVE_ASYNC_LDS
    async_ld_b128(g, Vn + row * 40 + c);
#else
    s.r = *(const v8bf*)g;
#endif
  }
}
__device__ inline void stagea_commit(bf16_t* Kn, bf16_t* Vn, int t, const StageA& s) {
#if HAVE_ASYNC_LDS
  (void)Kn; (void)Vn; (void)t; (void)s;
  wait_async0();
#else
  if (t < 128) {
    const int row = t >> 2, c = (t & 3) * 16;
    *(v8bf*)(Kn + row * 72 + c) = s.r;
  } else {
    const int tt = t - 128;
    const int row = tt >> 1, c = (tt & 1) * 16;
    *(v8bf*)(Vn + row * 40 + c) = s.r;
  }
#endif
}

__global__ void __launch_bounds__(256)
attention(const bf16_t* __restrict__ Qb, const bf16_t* __restrict__ Kb,
          const bf16_t* __restrict__ Vt, const int* __restrict__ mask,
          bf16_t* __restrict__ ctx) {
  __shared__ bf16_t Kt[2][32][72];
  __shared__ bf16_t Vtl[2][64][40];
  __shared__ bf16_t Pld[8][16][40];

  const int t = threadIdx.x;
  const int lane = t & 31, wave = t >> 5;
  const int h = blockIdx.y, b = blockIdx.z;
  const int q0 = (blockIdx.x * 8 + wave) * 16;
  const int bh = b * 16 + h;
  const bf16_t* Qp = Qb + (size_t)bh * 1024 * 64;
  const bf16_t* Kp = Kb + (size_t)bh * 1024 * 64;
  const bf16_t* Vp = Vt + (size_t)bh * 64 * 1024;
  const int*    mp = mask + b * 1024;

  const v16bf aq0 = load_afrag_g(Qp, 64, q0, 0,  lane);
  const v16bf aq1 = load_afrag_g(Qp, 64, q0, 32, lane);

  const int half = lane >> 4, cidx = lane & 15;
  float m[8], l[8];
#pragma unroll
  for (int v = 0; v < 8; ++v) { m[v] = -__builtin_inff(); l[v] = 0.f; }
  v8f acc[4] = {};
  bf16_t* Prow = &Pld[wave][0][0];    // row stride 40

  bf16_t* Kc = &Kt[0][0][0];  bf16_t* Vc = &Vtl[0][0][0];
  bf16_t* Kn = &Kt[1][0][0];  bf16_t* Vn = &Vtl[1][0][0];

  StageA sr;
  stagea_issue(Kp, Vp, 0, Kc, Vc, t, sr);
  stagea_commit(Kc, Vc, t, sr);
  __syncthreads();

#pragma unroll 1
  for (int kk = 0; kk < 32; ++kk) {
    const int kv0 = kk * 32;
    if (kk < 31) stagea_issue(Kp, Vp, kv0 + 32, Kn, Vn, t, sr);

    // ---- scores S (16x32) = Q @ K^T from LDS K tile ----
    v8f s[2] = {};
#pragma unroll
    for (int u = 0; u < 2; ++u) {
      v16bf bk0 = lds_bfrag(Kc, 72, 16 * u, 0,  lane);
      v16bf bk1 = lds_bfrag(Kc, 72, 16 * u, 32, lane);
      s[u] = wmma_bf16(aq0, bk0, s[u]);
      s[u] = wmma_bf16(aq1, bk1, s[u]);
    }
    // ---- padding mask ----
#pragma unroll
    for (int u = 0; u < 2; ++u) {
      if (mp[kv0 + 16 * u + cidx] == 0) {
#pragma unroll
        for (int v = 0; v < 8; ++v) s[u][v] = -__builtin_inff();
      }
    }
    // ---- online softmax (row = v + 8*half; reduce over 16 lanes) ----
    float scale[8];
#pragma unroll
    for (int v = 0; v < 8; ++v) {
      float rmax = fmaxf(s[0][v], s[1][v]);
      rmax = fmaxf(rmax, __shfl_xor(rmax, 1, 32));
      rmax = fmaxf(rmax, __shfl_xor(rmax, 2, 32));
      rmax = fmaxf(rmax, __shfl_xor(rmax, 4, 32));
      rmax = fmaxf(rmax, __shfl_xor(rmax, 8, 32));
      const float mnew = fmaxf(m[v], rmax);
      const float sc = __expf(m[v] - mnew);
      const float p0 = __expf(s[0][v] - mnew);
      const float p1 = __expf(s[1][v] - mnew);
      s[0][v] = p0; s[1][v] = p1;
      float rs = p0 + p1;
      rs += __shfl_xor(rs, 1, 32);
      rs += __shfl_xor(rs, 2, 32);
      rs += __shfl_xor(rs, 4, 32);
      rs += __shfl_xor(rs, 8, 32);
      l[v] = l[v] * sc + rs;
      m[v] = mnew;
      scale[v] = sc;
    }
#pragma unroll
    for (int j = 0; j < 4; ++j)
#pragma unroll
      for (int v = 0; v < 8; ++v)
        acc[j][v] *= scale[v];

    // ---- P (D layout) -> LDS -> A fragment (same wave; LDS is in-order) ----
#pragma unroll
    for (int u = 0; u < 2; ++u)
#pragma unroll
      for (int v = 0; v < 8; ++v)
        Prow[(v + 8 * half) * 40 + 16 * u + cidx] = f2bf(s[u][v]);
    const bf16_t* pp = Prow + (size_t)cidx * 40 + half * 8;
    v8bf plo = *(const v8bf*)pp;
    v8bf phi = *(const v8bf*)(pp + 16);
    const v16bf ap = join16(plo, phi);

    // ---- context += P(16x32) @ V(32x64) from LDS V tile ----
#pragma unroll
    for (int j = 0; j < 4; ++j) {
      v16bf bv = lds_bfrag(Vc, 40, 16 * j, 0, lane);
      acc[j] = wmma_bf16(ap, bv, acc[j]);
    }

    if (kk < 31) stagea_commit(Kn, Vn, t, sr);
    __syncthreads();
    bf16_t* tp;
    tp = Kc; Kc = Kn; Kn = tp;
    tp = Vc; Vc = Vn; Vn = tp;
  }

  // ---- finalize ----
#pragma unroll
  for (int v = 0; v < 8; ++v) l[v] = 1.0f / l[v];
#pragma unroll
  for (int j = 0; j < 4; ++j)
#pragma unroll
    for (int v = 0; v < 8; ++v) {
      const int row = q0 + v + 8 * half;
      const int dcol = h * 64 + 16 * j + cidx;
      ctx[((size_t)(b * 1024) + row) * 1024 + dcol] = f2bf(acc[j][v] * l[v]);
    }
}

// ================= output projection =================
__global__ void __launch_bounds__(256)
out_proj(const bf16_t* __restrict__ ctx, const bf16_t* __restrict__ wo,
         const float* __restrict__ bo, float* __restrict__ out) {
  __shared__ bf16_t Atile[2][128][40];
  __shared__ bf16_t Btile[2][256][40];

  const int t = threadIdx.x;
  const int lane = t & 31, wave = t >> 5;
  const int tmb = (blockIdx.x >> 2) * 128;
  const int tnb = (blockIdx.x & 3) * 256;
  const int wm = (wave >> 2) * 64;
  const int wn = (wave & 3) * 64;

  v8f acc[4][4] = {};
  gemm_core_tiled(ctx, wo, &Atile[0][0][0], &Btile[0][0][0], tmb, tnb, wm, wn, lane, t, acc);

  const int tm = tmb + wm, tn = tnb + wn;
  const int half = lane >> 4, c = lane & 15;
#pragma unroll
  for (int i = 0; i < 4; ++i)
#pragma unroll
    for (int j = 0; j < 4; ++j) {
      const int ncol = tn + 16 * j + c;
      const float bval = bo[ncol];
#pragma unroll
      for (int v = 0; v < 8; ++v) {
        const int mrow = tm + 16 * i + v + 8 * half;
        out[(size_t)mrow * 1024 + ncol] = acc[i][j][v] + bval;
      }
    }
}

// ================= host launcher =================
extern "C" void kernel_launch(void* const* d_in, const int* in_sizes, int n_in,
                              void* d_out, int out_size, void* d_ws, size_t ws_size,
                              hipStream_t stream) {
  (void)in_sizes; (void)n_in; (void)out_size; (void)ws_size;
  const float* x   = (const float*)d_in[0];
  const int*  mask = (const int*)d_in[1];
  const float* q_w = (const float*)d_in[2];
  const float* q_b = (const float*)d_in[3];
  const float* k_w = (const float*)d_in[4];
  const float* k_b = (const float*)d_in[5];
  const float* v_w = (const float*)d_in[6];
  const float* v_b = (const float*)d_in[7];
  const float* o_w = (const float*)d_in[8];
  const float* o_b = (const float*)d_in[9];
  float* out = (float*)d_out;

  bf16_t* xb  = (bf16_t*)d_ws;               // 4096*1024
  bf16_t* qwb = xb  + (size_t)4096 * 1024;
  bf16_t* kwb = qwb + (size_t)1024 * 1024;
  bf16_t* vwb = kwb + (size_t)1024 * 1024;
  bf16_t* owb = vwb + (size_t)1024 * 1024;
  bf16_t* Qb  = owb + (size_t)1024 * 1024;   // [b,h,s,dh]
  bf16_t* Kb  = Qb  + (size_t)4096 * 1024;   // [b,h,s,dh]
  bf16_t* Vt  = Kb  + (size_t)4096 * 1024;   // [b,h,dh,s]
  bf16_t* ctx = Vt  + (size_t)4096 * 1024;   // [b*s, dim]

  cvt_f32_to_bf16<<<4096, 256, 0, stream>>>(x,   xb,  4096 * 1024);
  cvt_f32_to_bf16<<<1024, 256, 0, stream>>>(q_w, qwb, 1024 * 1024);
  cvt_f32_to_bf16<<<1024, 256, 0, stream>>>(k_w, kwb, 1024 * 1024);
  cvt_f32_to_bf16<<<1024, 256, 0, stream>>>(v_w, vwb, 1024 * 1024);
  cvt_f32_to_bf16<<<1024, 256, 0, stream>>>(o_w, owb, 1024 * 1024);

  dim3 gproj(128, 1, 3);
  qkv_proj<<<gproj, 256, 0, stream>>>(xb, qwb, kwb, vwb, q_b, k_b, v_b, Qb, Kb, Vt);

  dim3 gattn(8, 16, 4);
  attention<<<gattn, 256, 0, stream>>>(Qb, Kb, Vt, mask, ctx);

  out_proj<<<128, 256, 0, stream>>>(ctx, owb, o_b, out);
}
